// SimCLoss2_37151467110995
// MI455X (gfx1250) — matrix operand order repaired and checked by reading, not compiled, not run on hardware
//
#include <hip/hip_runtime.h>

// SimCLR-style contrastive loss on MI455X (gfx1250, wave32, WMMA).
//
// closs_i = log( sum_{j!=i} exp(<z_i,z_j>/T) ) - <z_i,z'_i>/T ;  out = mean_i closs_i
// N=8192, D=256, T=0.05 (scale = 20).
//
// Dominant cost: 8192x8192x256 Gram matrix (34.4 GFLOP). z is 8MB -> L2/WGP$
// resident, so this is matrix-core bound. bf16x3 splitting (hi+lo) with
// v_wmma_f32_16x16x32_bf16: 3 WMMAs per K=32 chunk, ~fp32 accuracy.
//
//  * 3 independent WMMA accumulator chains (hh/hl/lh) -> 3-way XDL ILP
//  * B tiles staged into LDS once per workgroup via GLOBAL_LOAD_ASYNC_TO_LDS
//    (double-buffered, ASYNCcnt-tracked) instead of 8x redundant vmem reads;
//    builtin takes typed v4i pointers (AS1 src, AS3 dst) per hipcc diagnostic.

#define NROWS 8192
#define DDIM  256
#define KC    (DDIM / 32)      // 8 K-chunks of 32
#define TILES (NROWS / 16)     // 512 16-row/col tiles
#define CSPLIT 8               // column splits (grid.y)
#define CT_PER (TILES / CSPLIT)
#define INV_T 20.0f
#define FRAG_ELEMS (KC * 32 * 16)   // bf16 elements per (tile) fragment region = 4096

typedef __attribute__((ext_vector_type(16))) __bf16 v16bf;
typedef __attribute__((ext_vector_type(8)))  float  v8f;
typedef __attribute__((ext_vector_type(4)))  int    v4i;

#if defined(__has_builtin)
# if __has_builtin(__builtin_amdgcn_global_load_async_to_lds_b128)
#  define HAVE_ASYNC_LDS 1
# endif
#endif

#ifdef HAVE_ASYNC_LDS
# if __has_builtin(__builtin_amdgcn_s_wait_asynccnt)
#  define WAIT_ASYNC(n) __builtin_amdgcn_s_wait_asynccnt(n)
# else
#  define WAIT_ASYNC(n) asm volatile("s_wait_asynccnt %0" :: "i"(n) : "memory")
# endif
typedef __attribute__((address_space(1))) v4i gv4i;   // global int4
typedef __attribute__((address_space(3))) v4i lv4i;   // LDS int4
__device__ __forceinline__ void async_copy16(const void* g, void* l) {
    __builtin_amdgcn_global_load_async_to_lds_b128((gv4i*)g, (lv4i*)l, 0, 0);
}
#endif

__device__ __forceinline__ unsigned short f32_to_bf16_rn(float f) {
    unsigned int u = __float_as_uint(f);
    unsigned int r = u + 0x7FFFu + ((u >> 16) & 1u);   // round-to-nearest-even
    return (unsigned short)(r >> 16);
}
__device__ __forceinline__ float bf16_bits_to_f32(unsigned short h) {
    return __uint_as_float(((unsigned int)h) << 16);
}

// ---------------------------------------------------------------------------
// Pack z into WMMA fragment order (hi/lo split), for both A (16x32) and
// B (32x16) per-lane layouts, so the GEMM kernel loads fragments directly.
// A layout (16-bit, 16x32):  lanes 0-15: M=lane,   K = {0..7, 16..23}
//                            lanes16-31: M=lane-16,K = {8..15,24..31}
// B layout (16-bit, 32x16):  lanes 0-15: N=lane,   K = {0..7, 16..23}
//                            lanes16-31: N=lane-16,K = {8..15,24..31}
// Fragment element index: ((tile*KC + chunk)*32 + lane)*16 + j
// ---------------------------------------------------------------------------
__global__ __launch_bounds__(256)
void pack_frags(const float* __restrict__ z,
                unsigned short* __restrict__ Ah, unsigned short* __restrict__ Al,
                unsigned short* __restrict__ Bh, unsigned short* __restrict__ Bl)
{
    int id   = blockIdx.x * 256 + threadIdx.x;  // t*256 + c*32 + lane
    int lane = id & 31;
    int c    = (id >> 5) & (KC - 1);
    int t    = id >> 8;
    int laneLow = lane & 15;
    bool hiHalf = lane >= 16;
    int row = t * 16 + laneLow;                 // A-row; also B-column (= z row)
    long base = ((long)(t * KC + c) * 32 + lane) * 16;

    #pragma unroll
    for (int j = 0; j < 16; ++j) {
        // ---- A fragment element ----
        int koffA = hiHalf ? (j < 8 ? j + 8 : j + 16)
                           : (j < 8 ? j     : j + 8);
        float va = z[(long)row * DDIM + c * 32 + koffA];
        unsigned short ha = f32_to_bf16_rn(va);
        Ah[base + j] = ha;
        Al[base + j] = f32_to_bf16_rn(va - bf16_bits_to_f32(ha));
        // ---- B fragment element (B[k][n] = z[n][k]) ----
        int koffB = koffA;
        float vb = z[(long)row * DDIM + c * 32 + koffB];
        unsigned short hb = f32_to_bf16_rn(vb);
        Bh[base + j] = hb;
        Bl[base + j] = f32_to_bf16_rn(vb - bf16_bits_to_f32(hb));
    }
}

// ---------------------------------------------------------------------------
// Per-row positive-pair dot product (fp32) + zero-init of backsum.
// One wave (32 lanes) per row.
// ---------------------------------------------------------------------------
__global__ __launch_bounds__(256)
void pos_init(const float* __restrict__ z, const float* __restrict__ zp,
              float* __restrict__ posdot, float* __restrict__ backsum)
{
    int row  = (blockIdx.x * 256 + threadIdx.x) >> 5;
    int lane = threadIdx.x & 31;
    const float* a = z  + (long)row * DDIM;
    const float* b = zp + (long)row * DDIM;
    float s = 0.0f;
    #pragma unroll
    for (int k = lane; k < DDIM; k += 32) s += a[k] * b[k];
    #pragma unroll
    for (int m = 16; m >= 1; m >>= 1) s += __shfl_xor(s, m, 32);
    if (lane == 0) { posdot[row] = s; backsum[row] = 0.0f; }
}

// ---------------------------------------------------------------------------
// Main kernel: backsum[i] += sum_{j in col-split, j!=i} exp(20 * <z_i,z_j>)
// Wave = one 16-row strip; 8 waves per workgroup; grid = (TILES/8, CSPLIT).
// A fragments register-resident; B tiles double-buffered through LDS via
// async DMA (one copy per workgroup, shared by all 8 waves).
// ---------------------------------------------------------------------------
__global__ __launch_bounds__(256)
void gemm_exp_rowsum(const unsigned short* __restrict__ Ah,
                     const unsigned short* __restrict__ Al,
                     const unsigned short* __restrict__ Bh,
                     const unsigned short* __restrict__ Bl,
                     float* __restrict__ backsum)
{
    const int lane = threadIdx.x & 31;
    const int wave = threadIdx.x >> 5;
    const int tr   = blockIdx.x * 8 + wave;      // row-tile index
    const int ct0  = blockIdx.y * CT_PER;        // first col-tile of this split

    const int rowHalf   = (lane >> 4) << 3;      // 0 or 8 (C-layout row offset)
    const int colInTile = lane & 15;             // C-layout column

    const v16bf* pAh = reinterpret_cast<const v16bf*>(Ah);
    const v16bf* pAl = reinterpret_cast<const v16bf*>(Al);

    float rowAcc[8];
    #pragma unroll
    for (int m = 0; m < 8; ++m) rowAcc[m] = 0.0f;

#ifdef HAVE_ASYNC_LDS
    // [2 buffers][hi/lo][fragment region]  = 32 KB of the 320 KB LDS
    __shared__ __align__(16) unsigned short ldsB[2][2][FRAG_ELEMS];

    // stage tile ct's B fragments (hi+lo, 16 KB) into buffer b:
    // 256 threads x 4 async b128 each.
    auto stage = [&](int ct, int b) {
        const char* gh = (const char*)(Bh + (size_t)ct * FRAG_ELEMS);
        const char* gl = (const char*)(Bl + (size_t)ct * FRAG_ELEMS);
        char* lh = (char*)&ldsB[b][0][0];
        char* ll = (char*)&ldsB[b][1][0];
        int off = threadIdx.x * 16;
        async_copy16(gh + off,        lh + off);
        async_copy16(gl + off,        ll + off);
        async_copy16(gh + off + 4096, lh + off + 4096);
        async_copy16(gl + off + 4096, ll + off + 4096);
    };

    stage(ct0, 0);
    for (int j = 0; j < CT_PER; ++j) {
        const int b  = j & 1;
        const int ct = ct0 + j;
        if (j + 1 < CT_PER) {           // prefetch next tile into other buffer
            stage(ct + 1, b ^ 1);
            WAIT_ASYNC(4);              // tile j's 4 copies complete; next 4 in flight
        } else {
            WAIT_ASYNC(0);
        }
        __syncthreads();                // all waves' DMA visible

        const v16bf* lBh = (const v16bf*)&ldsB[b][0][0];
        const v16bf* lBl = (const v16bf*)&ldsB[b][1][0];

        v8f accHH = {}, accHL = {}, accLH = {};   // 3 independent XDL chains
        #pragma unroll
        for (int c = 0; c < KC; ++c) {
            const int ia = (tr * KC + c) * 32 + lane;
            const int ib = c * 32 + lane;
            v16bf ah = pAh[ia];
            v16bf al = pAl[ia];
            v16bf bh = lBh[ib];
            v16bf bl = lBl[ib];
            accHH = __builtin_amdgcn_wmma_f32_16x16x32_bf16(
                        false, ah, false, bh, (short)0, accHH, false, false);
            accHL = __builtin_amdgcn_wmma_f32_16x16x32_bf16(
                        false, ah, false, bl, (short)0, accHL, false, false);
            accLH = __builtin_amdgcn_wmma_f32_16x16x32_bf16(
                        false, al, false, bh, (short)0, accLH, false, false);
        }
        const bool diagTile = (tr == ct);
        #pragma unroll
        for (int m = 0; m < 8; ++m) {
            float s = accHH[m] + accHL[m] + accLH[m];
            float e = __expf(s * INV_T);
            if (diagTile && (rowHalf + m) == colInTile) e = 0.0f;  // mask j==i
            rowAcc[m] += e;
        }
        __syncthreads();                // done reading buffer b before restage
    }
#else
    // Fallback: direct global loads of B fragments (round-1 scheme, 3 chains).
    const v16bf* pBh = reinterpret_cast<const v16bf*>(Bh);
    const v16bf* pBl = reinterpret_cast<const v16bf*>(Bl);
    for (int j = 0; j < CT_PER; ++j) {
        const int ct = ct0 + j;
        v8f accHH = {}, accHL = {}, accLH = {};
        #pragma unroll
        for (int c = 0; c < KC; ++c) {
            const int ia = (tr * KC + c) * 32 + lane;
            const int ib = (ct * KC + c) * 32 + lane;
            v16bf ah = pAh[ia];
            v16bf al = pAl[ia];
            v16bf bh = pBh[ib];
            v16bf bl = pBl[ib];
            accHH = __builtin_amdgcn_wmma_f32_16x16x32_bf16(
                        false, ah, false, bh, (short)0, accHH, false, false);
            accHL = __builtin_amdgcn_wmma_f32_16x16x32_bf16(
                        false, ah, false, bl, (short)0, accHL, false, false);
            accLH = __builtin_amdgcn_wmma_f32_16x16x32_bf16(
                        false, al, false, bh, (short)0, accLH, false, false);
        }
        const bool diagTile = (tr == ct);
        #pragma unroll
        for (int m = 0; m < 8; ++m) {
            float s = accHH[m] + accHL[m] + accLH[m];
            float e = __expf(s * INV_T);
            if (diagTile && (rowHalf + m) == colInTile) e = 0.0f;
            rowAcc[m] += e;
        }
    }
#endif

    // Reduce the 16 lanes of each half (they hold distinct columns of one row).
    #pragma unroll
    for (int m = 0; m < 8; ++m) {
        float v = rowAcc[m];
        v += __shfl_xor(v, 1, 32);
        v += __shfl_xor(v, 2, 32);
        v += __shfl_xor(v, 4, 32);
        v += __shfl_xor(v, 8, 32);
        rowAcc[m] = v;
    }
    if (colInTile == 0) {                        // lanes 0 (rows m) and 16 (rows m+8)
        const int r0 = tr * 16 + rowHalf;
        #pragma unroll
        for (int m = 0; m < 8; ++m)
            atomicAdd(&backsum[r0 + m], rowAcc[m]);
    }
}

// ---------------------------------------------------------------------------
// out = mean_i( log(backsum[i]) - 20 * posdot[i] )
// ---------------------------------------------------------------------------
__global__ __launch_bounds__(256)
void finalize(const float* __restrict__ backsum, const float* __restrict__ posdot,
              float* __restrict__ out)
{
    __shared__ float sm[256];
    float s = 0.0f;
    for (int i = threadIdx.x; i < NROWS; i += 256)
        s += __logf(backsum[i]) - posdot[i] * INV_T;
    sm[threadIdx.x] = s;
    __syncthreads();
    #pragma unroll
    for (int k = 128; k > 0; k >>= 1) {
        if (threadIdx.x < k) sm[threadIdx.x] += sm[threadIdx.x + k];
        __syncthreads();
    }
    if (threadIdx.x == 0) out[0] = sm[0] / (float)NROWS;
}

extern "C" void kernel_launch(void* const* d_in, const int* in_sizes, int n_in,
                              void* d_out, int out_size, void* d_ws, size_t ws_size,
                              hipStream_t stream) {
    (void)in_sizes; (void)n_in; (void)out_size; (void)ws_size;
    const float* z  = (const float*)d_in[0];
    const float* zp = (const float*)d_in[1];
    float* out = (float*)d_out;

    char* ws = (char*)d_ws;
    const size_t PACK_BYTES = (size_t)NROWS * DDIM * 2;   // 4 MB each
    unsigned short* Ah = (unsigned short*)(ws);
    unsigned short* Al = (unsigned short*)(ws + 1 * PACK_BYTES);
    unsigned short* Bh = (unsigned short*)(ws + 2 * PACK_BYTES);
    unsigned short* Bl = (unsigned short*)(ws + 3 * PACK_BYTES);
    float* backsum = (float*)(ws + 4 * PACK_BYTES);
    float* posdot  = (float*)(ws + 4 * PACK_BYTES + NROWS * sizeof(float));

    // 1) pack fragments: TILES*KC*32 lanes = 131072 threads
    pack_frags<<<TILES, 256, 0, stream>>>(z, Ah, Al, Bh, Bl);
    // 2) positive-pair dots + zero backsum: one wave per row
    pos_init<<<(NROWS * 32) / 256, 256, 0, stream>>>(z, zp, posdot, backsum);
    // 3) big WMMA GEMM + exp row sums
    gemm_exp_rowsum<<<dim3(TILES / 8, CSPLIT), 256, 0, stream>>>(Ah, Al, Bh, Bl, backsum);
    // 4) final scalar
    finalize<<<1, 256, 0, stream>>>(backsum, posdot, out);
}
